// RLAPv3_36773509989005
// MI455X (gfx1250) — matrix operand order repaired and checked by reading, not compiled, not run on hardware
//
#include <hip/hip_runtime.h>
#include <hip/hip_bf16.h>
#include <math.h>

// ---- problem constants (match reference) ----
#define BB 128
#define CC 576
#define HH 28
#define WW 28
#define HW 784
#define KPAD 800          // 784 padded to multiple of 32 for WMMA K
#define NA 6              // angles
#define TPB 16            // (b,c) tiles per block in pass 1
#define NTILES (BB*CC)    // 73728

typedef _Float16 v16h __attribute__((ext_vector_type(16)));
typedef float    v8f  __attribute__((ext_vector_type(8)));
typedef _Float16 h8   __attribute__((ext_vector_type(8)));

__device__ __forceinline__ float sigm(float x) {
    return 1.0f / (1.0f + __expf(-x));
}

// =====================================================================
// Pass 1: per-(b,c)-tile statistics + orientation GEMM (WMMA).
// grid = NTILES/TPB blocks of 256 threads (8 waves). Each block owns 16
// consecutive (b,c) tiles (16 x 784 contiguous floats).
// =====================================================================
__global__ __launch_bounds__(256)
void rlap_k1(const float* __restrict__ x,
             const float* __restrict__ h_w, const float* __restrict__ v_w,
             const float* __restrict__ f_w,
             float* __restrict__ yh,   // [NTILES, H]
             float* __restrict__ yv,   // [NTILES, W]
             float* __restrict__ At,   // [NTILES]
             float* __restrict__ stats /* [6*CC]: hS,hS2,vS,vS2,fS,fS2 */) {
    __shared__ __align__(16) _Float16 xh[TPB][KPAD];   // tiles as f16 (A matrix)
    __shared__ __align__(16) _Float16 mh[NA][KPAD];    // angle masks as f16 (B matrix)
    __shared__ __align__(16) _Float16 tmx[8][HW];      // per-wave horiz-max scratch
    __shared__ float pw[8][64];                        // per-wave ph[28] / pv[28]
    __shared__ float respL[16][16];                    // GEMM result scatter

    const int tid  = threadIdx.x;
    const int lane = tid & 31;
    const int wv   = tid >> 5;
    const long tile0 = (long)blockIdx.x * TPB;

    // ---- stage 16 tiles -> f16 LDS ----
    const float* xb = x + tile0 * HW;
    for (int i = tid; i < TPB * HW; i += 256) {
        int m = i / HW, k = i % HW;
        xh[m][k] = (_Float16)xb[i];
    }
    // zero K padding (784..799) : exactly 16*16 = 256 elements
    {
        int m = tid / (KPAD - HW);
        int k = HW + (tid % (KPAD - HW));
        xh[m][k] = (_Float16)0.0f;
    }
    // ---- angle masks (computed in-kernel; cheap) ----
    for (int j = tid; j < NA * KPAD; j += 256) {
        int a = j / KPAD, k = j % KPAD;
        _Float16 mval = (_Float16)0.0f;
        if (k < HW) {
            int h = k / WW, w = k % WW;
            float yy = -1.0f + 2.0f * (float)h / (float)(HH - 1);
            float xx = -1.0f + 2.0f * (float)w / (float)(WW - 1);
            const float ang[NA] = {0.f, 30.f, 45.f, 60.f, 90.f, 135.f};
            float t = ang[a] * 0.017453292519943295f;
            float perp = xx * (-__sinf(t)) + yy * __cosf(t);
            float p = perp * (1.0f / 0.3f);
            mval = (_Float16)__expf(-0.5f * p * p);
        }
        mh[a][k] = mval;
    }
    __syncthreads();

    // ---- wave 0: orientation responses via V_WMMA_F32_16X16X32_F16 ----
    // D[m][n] = sum_k xh[m][k] * mh[n][k]   (m = tile in block, n = angle)
    if (wv == 0) {
        const int half = lane >> 4;   // 0: lanes 0-15, 1: lanes 16-31
        const int q    = lane & 15;   // M for A, N for B
        v8f acc = {};
        for (int kc = 0; kc < KPAD / 32; ++kc) {
            // A fragment: lane<16 -> K 0..7 & 16..23 ; lane>=16 -> K 8..15 & 24..31
            int ka = kc * 32 + half * 8;
            h8 a0 = *(const h8*)&xh[q][ka];
            h8 a1 = *(const h8*)&xh[q][ka + 16];
            v16h Af;
            #pragma unroll
            for (int i = 0; i < 8; ++i) { Af[i] = a0[i]; Af[i + 8] = a1[i]; }
            // B fragment: lane<16 -> K 0..7 & 16..23 ; lane>=16 -> K 8..15 & 24..31
            int kb = kc * 32 + half * 8;
            v16h Bf;
            if (q < NA) {
                h8 b0 = *(const h8*)&mh[q][kb];
                h8 b1 = *(const h8*)&mh[q][kb + 16];
                #pragma unroll
                for (int i = 0; i < 8; ++i) { Bf[i] = b0[i]; Bf[i + 8] = b1[i]; }
            } else {
                #pragma unroll
                for (int i = 0; i < 16; ++i) Bf[i] = (_Float16)0.0f;
            }
            acc = __builtin_amdgcn_wmma_f32_16x16x32_f16(
                false, Af, false, Bf, (short)0, acc, false, false);
        }
        // D layout: VGPR r holds M=r (lanes 0-15) / M=8+r (lanes 16-31), N = q
        #pragma unroll
        for (int r = 0; r < 8; ++r) respL[8 * half + r][q] = acc[r];
        // same-wave LDS is in-order: safe to read back immediately
        if (lane < 16) {
            float s = 0.0f;
            #pragma unroll
            for (int a = 0; a < NA; ++a)
                s += sigm(respL[lane][a] * (1.0f / (float)HW));
            At[tile0 + lane] = s * (1.0f / (float)NA);
        }
    }

    // ---- all waves: pooled streams + maxpool stats, 2 tiles per wave ----
    for (int s = 0; s < 2; ++s) {
        const int m  = wv * 2 + s;
        const long tg = tile0 + m;
        const int c  = (int)(tg % CC);

        // row means (over W) and column means (over H)
        if (lane < HH) {
            float acc = 0.0f;
            #pragma unroll 4
            for (int w = 0; w < WW; ++w) acc += (float)xh[m][lane * WW + w];
            pw[wv][lane] = acc * (1.0f / (float)WW);
        }
        if (lane < WW) {
            float acc = 0.0f;
            #pragma unroll 4
            for (int h = 0; h < HH; ++h) acc += (float)xh[m][h * WW + lane];
            pw[wv][32 + lane] = acc * (1.0f / (float)HH);
        }
        // K=3 depthwise conv over H (cross-correlation, zero 'SAME' pad)
        {
            float w0 = h_w[c * 3 + 0], w1 = h_w[c * 3 + 1], w2 = h_w[c * 3 + 2];
            float val = 0.0f;
            if (lane < HH) {
                float a = (lane > 0)      ? pw[wv][lane - 1] : 0.0f;
                float b = pw[wv][lane];
                float d = (lane < HH - 1) ? pw[wv][lane + 1] : 0.0f;
                val = a * w0 + b * w1 + d * w2;
                yh[tg * HH + lane] = val;
            }
            float s1 = (lane < HH) ? val : 0.0f;
            float s2 = s1 * s1;
            for (int o = 16; o; o >>= 1) {
                s1 += __shfl_xor(s1, o, 32);
                s2 += __shfl_xor(s2, o, 32);
            }
            if (lane == 0) {
                atomicAdd(&stats[0 * CC + c], s1);
                atomicAdd(&stats[1 * CC + c], s2);
            }
        }
        // K=3 depthwise conv over W
        {
            float w0 = v_w[c * 3 + 0], w1 = v_w[c * 3 + 1], w2 = v_w[c * 3 + 2];
            float val = 0.0f;
            if (lane < WW) {
                float a = (lane > 0)      ? pw[wv][32 + lane - 1] : 0.0f;
                float b = pw[wv][32 + lane];
                float d = (lane < WW - 1) ? pw[wv][32 + lane + 1] : 0.0f;
                val = a * w0 + b * w1 + d * w2;
                yv[tg * WW + lane] = val;
            }
            float s1 = (lane < WW) ? val : 0.0f;
            float s2 = s1 * s1;
            for (int o = 16; o; o >>= 1) {
                s1 += __shfl_xor(s1, o, 32);
                s2 += __shfl_xor(s2, o, 32);
            }
            if (lane == 0) {
                atomicAdd(&stats[2 * CC + c], s1);
                atomicAdd(&stats[3 * CC + c], s2);
            }
        }
        // separable 7x7 maxpool ('SAME', -inf pad) -> white top-hat stats
        for (int idx = lane; idx < HW; idx += 32) {
            int h = idx / WW, w = idx % WW;
            int wlo = (w - 3 > 0) ? w - 3 : 0;
            int whi = (w + 3 < WW - 1) ? w + 3 : WW - 1;
            float mx = -3.0e38f;
            for (int ww2 = wlo; ww2 <= whi; ++ww2)
                mx = fmaxf(mx, (float)xh[m][h * WW + ww2]);
            tmx[wv][idx] = (_Float16)mx;
        }
        {
            float fw = f_w[c];
            float fs1 = 0.0f, fs2 = 0.0f;
            for (int idx = lane; idx < HW; idx += 32) {
                int h = idx / WW, w = idx % WW;
                int hlo = (h - 3 > 0) ? h - 3 : 0;
                int hhi = (h + 3 < HH - 1) ? h + 3 : HH - 1;
                float mx = -3.0e38f;
                for (int hh2 = hlo; hh2 <= hhi; ++hh2)
                    mx = fmaxf(mx, (float)tmx[wv][hh2 * WW + w]);
                float sp = (mx - (float)xh[m][idx]) * fw;
                fs1 += sp; fs2 += sp * sp;
            }
            for (int o = 16; o; o >>= 1) {
                fs1 += __shfl_xor(fs1, o, 32);
                fs2 += __shfl_xor(fs2, o, 32);
            }
            if (lane == 0) {
                atomicAdd(&stats[4 * CC + c], fs1);
                atomicAdd(&stats[5 * CC + c], fs2);
            }
        }
    }
}

// =====================================================================
// Pass 2: fold BN stats + gamma/beta (+ f_w) into per-channel scale/off
// =====================================================================
__global__ void rlap_k2(const float* __restrict__ stats,
                        const float* __restrict__ h_g, const float* __restrict__ h_b,
                        const float* __restrict__ v_g, const float* __restrict__ v_b,
                        const float* __restrict__ f_g, const float* __restrict__ f_b,
                        const float* __restrict__ f_w,
                        float* __restrict__ coef /* [6*CC] */) {
    int c = blockIdx.x * blockDim.x + threadIdx.x;
    if (c >= CC) return;
    const float invNH = 1.0f / (float)(BB * HH);
    const float invNF = 1.0f / ((float)BB * HH * WW);
    const float eps = 1e-5f;

    float m = stats[0 * CC + c] * invNH;
    float v = stats[1 * CC + c] * invNH - m * m;
    float sc = h_g[c] * rsqrtf(v + eps);
    coef[0 * CC + c] = sc;
    coef[1 * CC + c] = h_b[c] - m * sc;

    m = stats[2 * CC + c] * invNH;
    v = stats[3 * CC + c] * invNH - m * m;
    sc = v_g[c] * rsqrtf(v + eps);
    coef[2 * CC + c] = sc;
    coef[3 * CC + c] = v_b[c] - m * sc;

    m = stats[4 * CC + c] * invNF;         // stats are of s = spots*f_w
    v = stats[5 * CC + c] * invNF - m * m;
    float r = f_g[c] * rsqrtf(v + eps);
    coef[4 * CC + c] = f_w[c] * r;          // applied directly to raw spots
    coef[5 * CC + c] = f_b[c] - m * r;
}

// =====================================================================
// Pass 3: apply the four gates. One block (256 thr) per (b,c) tile.
// =====================================================================
__global__ __launch_bounds__(256)
void rlap_k3(const float* __restrict__ x,
             const float* __restrict__ yh, const float* __restrict__ yv,
             const float* __restrict__ At, const float* __restrict__ coef,
             float* __restrict__ out) {
    __shared__ float xs[HW];
    __shared__ float tmx[HW];
    __shared__ float ah[HH];
    __shared__ float av[WW];
    __shared__ float at_s;

    const long tg = blockIdx.x;
    const int c = (int)(tg % CC);
    const int tid = threadIdx.x;
    const float* xb = x + tg * HW;

    for (int i = tid; i < HW; i += 256) xs[i] = xb[i];
    if (tid == 0) at_s = At[tg];
    __syncthreads();

    if (tid < HH)
        ah[tid] = sigm(yh[tg * HH + tid] * coef[0 * CC + c] + coef[1 * CC + c]);
    else if (tid >= 32 && tid < 32 + WW) {
        int w = tid - 32;
        av[w] = sigm(yv[tg * WW + w] * coef[2 * CC + c] + coef[3 * CC + c]);
    }
    for (int i = tid; i < HW; i += 256) {
        int h = i / WW, w = i % WW;
        int wlo = (w - 3 > 0) ? w - 3 : 0;
        int whi = (w + 3 < WW - 1) ? w + 3 : WW - 1;
        float mx = -3.0e38f;
        for (int ww2 = wlo; ww2 <= whi; ++ww2)
            mx = fmaxf(mx, xs[h * WW + ww2]);
        tmx[i] = mx;
    }
    __syncthreads();

    const float fsc = coef[4 * CC + c], fof = coef[5 * CC + c], at = at_s;
    for (int i = tid; i < HW; i += 256) {
        int h = i / WW, w = i % WW;
        int hlo = (h - 3 > 0) ? h - 3 : 0;
        int hhi = (h + 3 < HH - 1) ? h + 3 : HH - 1;
        float mx = -3.0e38f;
        for (int hh2 = hlo; hh2 <= hhi; ++hh2)
            mx = fmaxf(mx, tmx[hh2 * WW + w]);
        float sp = mx - xs[i];
        float af = sigm(sp * fsc + fof);
        out[tg * HW + i] = xs[i] * ah[h] * av[w] * af * at;
    }
}

// =====================================================================
extern "C" void kernel_launch(void* const* d_in, const int* in_sizes, int n_in,
                              void* d_out, int out_size, void* d_ws, size_t ws_size,
                              hipStream_t stream) {
    const float* x   = (const float*)d_in[0];
    const float* h_w = (const float*)d_in[1];
    const float* h_g = (const float*)d_in[2];
    const float* h_b = (const float*)d_in[3];
    const float* v_w = (const float*)d_in[4];
    const float* v_g = (const float*)d_in[5];
    const float* v_b = (const float*)d_in[6];
    const float* f_w = (const float*)d_in[7];
    const float* f_g = (const float*)d_in[8];
    const float* f_b = (const float*)d_in[9];
    float* out = (float*)d_out;

    // workspace layout (floats)
    float* ws    = (float*)d_ws;
    float* yh    = ws;                          // NTILES*HH
    float* yv    = yh + (size_t)NTILES * HH;    // NTILES*WW
    float* At    = yv + (size_t)NTILES * WW;    // NTILES
    float* stats = At + (size_t)NTILES;         // 6*CC
    float* coef  = stats + 6 * CC;              // 6*CC

    hipMemsetAsync(stats, 0, 6 * CC * sizeof(float), stream);

    rlap_k1<<<NTILES / TPB, 256, 0, stream>>>(x, h_w, v_w, f_w, yh, yv, At, stats);
    rlap_k2<<<(CC + 255) / 256, 256, 0, stream>>>(stats, h_g, h_b, v_g, v_b,
                                                  f_g, f_b, f_w, coef);
    rlap_k3<<<NTILES, 256, 0, stream>>>(x, yh, yv, At, coef, out);
}
